// DecoderAttention_77335181131845
// MI455X (gfx1250) — matrix operand order repaired and checked by reading, not compiled, not run on hardware
//
#include <hip/hip_runtime.h>

// ---------------------------------------------------------------------------
// DecoderAttention: B=2, T=2048, C=1024, H=16, D=64, causal, fp32 in/out.
// All GEMM stages on v_wmma_f32_16x16x32_bf16 (fp32 accumulate), flash
// attention (no TxT materialization), fp32 softmax.
// v3: V-transpose load phase now uses GLOBAL_LOAD_ASYNC_TO_LDS_B128
//     (ASYNCcnt-tracked DMA into LDS, no VGPR round-trip).
// ---------------------------------------------------------------------------

#define BB 2
#define TT 2048
#define CC 1024
#define HH 16
#define DD 64
#define MM (BB * TT)          // 4096 rows

typedef __attribute__((ext_vector_type(16))) __bf16 v16bf;
typedef __attribute__((ext_vector_type(8)))  __bf16 v8bf;
typedef __attribute__((ext_vector_type(8)))  float  v8f;

union V16U { v16bf v; v8bf h[2]; };

// branchless RNE float->bf16 (inputs are finite; no NaN path -> no EXEC churn)
static __device__ __forceinline__ unsigned short f32_to_bf16(float f) {
  unsigned int u = __float_as_uint(f);
  return (unsigned short)((u + 0x7fffu + ((u >> 16) & 1u)) >> 16);
}

static __device__ __forceinline__ v8f wmma_bf16(v16bf a, v16bf b, v8f c) {
  return __builtin_amdgcn_wmma_f32_16x16x32_bf16(false, a, false, b, (short)0, c,
                                                 false, false);
}

// ---- fragment loaders (CDNA5 ISA 7.12.2, wave32) ---------------------------
// A-matrix 16x32 bf16, row-major source, leading dim `ld` (elements).
static __device__ __forceinline__ v16bf load_a_frag(const unsigned short* base, int ld) {
  int lane = threadIdx.x & 31;
  int half = lane >> 4, m = lane & 15;
  const unsigned short* row = base + m * ld + half * 16;
  V16U u;
  u.h[0] = *(const v8bf*)(row);
  u.h[1] = *(const v8bf*)(row + 8);
  return u.v;
}

// B-matrix 32x16 bf16 with K contiguous in memory:
// element (K = half*16 + i, N = lane&15) at base + N*ld + K.
static __device__ __forceinline__ v16bf load_b_frag_kcontig(const unsigned short* base,
                                                            size_t ld) {
  int lane = threadIdx.x & 31;
  int half = lane >> 4, n = lane & 15;
  const unsigned short* p = base + (size_t)n * ld + half * 16;
  V16U u;
  u.h[0] = *(const v8bf*)(p);
  u.h[1] = *(const v8bf*)(p + 8);
  return u.v;
}

// ---------------------------------------------------------------------------
// Kernel 0: fp32 -> bf16 convert, vectorized x4
// ---------------------------------------------------------------------------
__global__ void cvt_f32_bf16_kernel(const float4* __restrict__ src,
                                    ushort4* __restrict__ dst, int n4) {
  for (int i = blockIdx.x * blockDim.x + threadIdx.x; i < n4; i += gridDim.x * blockDim.x) {
    float4 f = src[i];
    ushort4 o;
    o.x = f32_to_bf16(f.x); o.y = f32_to_bf16(f.y);
    o.z = f32_to_bf16(f.z); o.w = f32_to_bf16(f.w);
    dst[i] = o;
  }
}

// ---------------------------------------------------------------------------
// Double-buffered GEMM strip: one wave computes 32 rows x 64 cols of
// Y = A @ W^T, K = CC.  8 WMMAs per K-step; next K-step's fragments are
// issued before the current WMMAs so s_wait overlaps compute.
// ---------------------------------------------------------------------------
#define GEMM_LOAD_SET(a0, a1, b0, b1, b2, b3, kk)                               \
  a0 = load_a_frag(Abase + (kk), CC);                                           \
  a1 = load_a_frag(Abase + 16 * CC + (kk), CC);                                 \
  b0 = load_b_frag_kcontig(Wbase + 0 * 16 * CC + (kk), CC);                     \
  b1 = load_b_frag_kcontig(Wbase + 1 * 16 * CC + (kk), CC);                     \
  b2 = load_b_frag_kcontig(Wbase + 2 * 16 * CC + (kk), CC);                     \
  b3 = load_b_frag_kcontig(Wbase + 3 * 16 * CC + (kk), CC);

#define GEMM_WMMA_SET(a0, a1, b0, b1, b2, b3)                                   \
  acc[0][0] = wmma_bf16(a0, b0, acc[0][0]);                                     \
  acc[0][1] = wmma_bf16(a0, b1, acc[0][1]);                                     \
  acc[0][2] = wmma_bf16(a0, b2, acc[0][2]);                                     \
  acc[0][3] = wmma_bf16(a0, b3, acc[0][3]);                                     \
  acc[1][0] = wmma_bf16(a1, b0, acc[1][0]);                                     \
  acc[1][1] = wmma_bf16(a1, b1, acc[1][1]);                                     \
  acc[1][2] = wmma_bf16(a1, b2, acc[1][2]);                                     \
  acc[1][3] = wmma_bf16(a1, b3, acc[1][3]);

static __device__ __forceinline__ void gemm_strip_32x64(
    const unsigned short* __restrict__ Abase,   // row m0, col 0 (ld = CC)
    const unsigned short* __restrict__ Wbase,   // weight row n0 (ld = CC)
    v8f acc[2][4]) {
  v16bf xa0, xa1, xb0, xb1, xb2, xb3;           // buffer X
  v16bf ya0, ya1, yb0, yb1, yb2, yb3;           // buffer Y
  GEMM_LOAD_SET(xa0, xa1, xb0, xb1, xb2, xb3, 0)
  for (int k0 = 0; k0 < CC; k0 += 64) {
    GEMM_LOAD_SET(ya0, ya1, yb0, yb1, yb2, yb3, k0 + 32)
    GEMM_WMMA_SET(xa0, xa1, xb0, xb1, xb2, xb3)
    const int kn = (k0 + 64) & (CC - 1);        // wraps to 0 on last iter (unused)
    GEMM_LOAD_SET(xa0, xa1, xb0, xb1, xb2, xb3, kn)
    GEMM_WMMA_SET(ya0, ya1, yb0, yb1, yb2, yb3)
  }
}

// ---------------------------------------------------------------------------
// Kernel 1: fused QKV projection -> bf16 [B,H,T,D].
// Block 256 (8 waves); wave w owns rows m0 = bx*256 + w*32.
// grid = (M/256, C/64, 3)
// ---------------------------------------------------------------------------
__global__ __launch_bounds__(256) void qkv_proj_kernel(
    const unsigned short* __restrict__ xb,
    const unsigned short* __restrict__ wq, const unsigned short* __restrict__ wk,
    const unsigned short* __restrict__ wv,
    const float* __restrict__ bq, const float* __restrict__ bk,
    const float* __restrict__ bv,
    unsigned short* __restrict__ Qb, unsigned short* __restrict__ Kb,
    unsigned short* __restrict__ Vb) {
  const int wave = threadIdx.x >> 5;
  const int lane = threadIdx.x & 31;
  const int half = lane >> 4, nl = lane & 15;
  const int m0 = blockIdx.x * 256 + wave * 32;
  const int n0 = blockIdx.y * 64;
  const int mat = blockIdx.z;
  const unsigned short* W = (mat == 0) ? wq : (mat == 1) ? wk : wv;
  const float* bias        = (mat == 0) ? bq : (mat == 1) ? bk : bv;
  unsigned short* out      = (mat == 0) ? Qb : (mat == 1) ? Kb : Vb;

  v8f acc[2][4];
#pragma unroll
  for (int r = 0; r < 2; ++r)
#pragma unroll
    for (int n = 0; n < 4; ++n)
#pragma unroll
      for (int j = 0; j < 8; ++j) acc[r][n][j] = 0.0f;

  gemm_strip_32x64(xb + (size_t)m0 * CC, W + (size_t)n0 * CC, acc);

#pragma unroll
  for (int n = 0; n < 4; ++n) {
    const int ncol = n0 + n * 16 + nl;
    const int h = ncol >> 6, d = ncol & 63;
    const float bv_ = bias[ncol];
#pragma unroll
    for (int r = 0; r < 2; ++r)
#pragma unroll
      for (int j = 0; j < 8; ++j) {
        const int row = m0 + r * 16 + half * 8 + j;     // [0, B*T)
        const int bidx = row >> 11, t = row & (TT - 1);
        out[(((size_t)(bidx * HH + h) * TT + t) * DD) + d] =
            f32_to_bf16(acc[r][n][j] + bv_);
      }
  }
}

// ---------------------------------------------------------------------------
// Kernel 2: V transpose [B,H,T,D] -> [B,H,D,T].
// Load phase: GLOBAL_LOAD_ASYNC_TO_LDS_B128 (no VGPR round-trip, ASYNCcnt);
// each wave waits on its own async transfers, then a block barrier publishes
// the tile.  Row stride 72 ushorts = 144 B (16 B multiple for aligned b128
// async writes, and non-power-of-two to spread transposed column reads).
// grid = (T/64, B*H), block = 256.
// ---------------------------------------------------------------------------
__global__ __launch_bounds__(256) void v_transpose_kernel(
    const unsigned short* __restrict__ Vb, unsigned short* __restrict__ Vt) {
  __shared__ unsigned short tile[64][72];
  const int bh = blockIdx.y;
  const int t0 = blockIdx.x * 64;
  const unsigned short* src = Vb + (size_t)bh * TT * DD + (size_t)t0 * DD;
  const int tid = threadIdx.x;

  // generic->LDS offset: low 32 bits of a generic LDS address are the LDS addr
  const unsigned lds_base = (unsigned)(unsigned long long)(uintptr_t)&tile[0][0];
#pragma unroll
  for (int i = 0; i < 2; ++i) {
    const int lin = i * 256 + tid;               // unit = 8 ushorts (16 B)
    const int r = lin >> 3, c8 = lin & 7;
    const unsigned lds_off = lds_base + (unsigned)((r * 72 + c8 * 8) * 2);
    const unsigned long long gaddr =
        (unsigned long long)(uintptr_t)(src + r * DD + c8 * 8);
    asm volatile("global_load_async_to_lds_b128 %0, %1, off"
                 :: "v"(lds_off), "v"(gaddr)
                 : "memory");
  }
  asm volatile("s_wait_asynccnt 0x0" ::: "memory");
  __syncthreads();

  unsigned short* dst = Vt + (size_t)bh * DD * TT + t0;
#pragma unroll
  for (int i = 0; i < 4; ++i) {
    const int lin = i * 256 + tid;
    const int d = lin >> 4, c4 = lin & 15;
    ushort4 v;
    v.x = tile[c4 * 4 + 0][d]; v.y = tile[c4 * 4 + 1][d];
    v.z = tile[c4 * 4 + 2][d]; v.w = tile[c4 * 4 + 3][d];
    *(ushort4*)(dst + (size_t)d * TT + c4 * 4) = v;
  }
}

// ---------------------------------------------------------------------------
// Kernel 3: flash attention, causal.  Block 128 (4 waves); each wave owns 16
// query rows and streams 64-key blocks.  grid = (T/64, B*H).
// Q,K bf16 [B,H,T,D]; Vt bf16 [B,H,D,T]; out bf16 head-merged [B*T, C].
// ---------------------------------------------------------------------------
__global__ __launch_bounds__(128) void flash_attn_kernel(
    const unsigned short* __restrict__ Qb, const unsigned short* __restrict__ Kb,
    const unsigned short* __restrict__ Vt, unsigned short* __restrict__ yb) {
  __shared__ unsigned short sP[4][16 * 64];      // per-wave P tile (16x64 bf16)

  const int wave = threadIdx.x >> 5;
  const int lane = threadIdx.x & 31;
  const int half = lane >> 4, nl = lane & 15;
  const int bh = blockIdx.y;                     // b*H + h
  const int b = bh >> 4, h = bh & (HH - 1);
  const int q0 = blockIdx.x * 64 + wave * 16;

  const unsigned short* Qh = Qb + (size_t)bh * TT * DD;
  const unsigned short* Kh = Kb + (size_t)bh * TT * DD;
  const unsigned short* Vh = Vt + (size_t)bh * DD * TT;
  unsigned short* sp = &sP[wave][0];

  const v16bf qa0 = load_a_frag(Qh + (size_t)q0 * DD + 0, DD);
  const v16bf qa1 = load_a_frag(Qh + (size_t)q0 * DD + 32, DD);

  v8f accO[4];
  float mrow[8], lrow[8];
#pragma unroll
  for (int j = 0; j < 8; ++j) { mrow[j] = -INFINITY; lrow[j] = 0.0f; }
#pragma unroll
  for (int n = 0; n < 4; ++n)
#pragma unroll
    for (int j = 0; j < 8; ++j) accO[n][j] = 0.0f;

  constexpr float ss = 0.18033688011112042f;     // (1/sqrt(64)) * log2(e)

  for (int k0 = 0; k0 < q0 + 16; k0 += 64) {
    // ---- S = Q K^T over 64 keys: 4 score frags, K-dim = D in 2 steps ------
    v8f S[4];
#pragma unroll
    for (int kk = 0; kk < 4; ++kk)
#pragma unroll
      for (int j = 0; j < 8; ++j) S[kk][j] = 0.0f;
#pragma unroll
    for (int ds = 0; ds < 2; ++ds) {
      const v16bf qa = ds ? qa1 : qa0;
#pragma unroll
      for (int kk = 0; kk < 4; ++kk) {
        v16bf kb = load_b_frag_kcontig(Kh + (size_t)(k0 + kk * 16) * DD + ds * 32, DD);
        S[kk] = wmma_bf16(qa, kb, S[kk]);
      }
    }

    // ---- prefetch V fragments now; latency hides behind softmax VALU ------
    v16bf vb[2][4];
#pragma unroll
    for (int ks = 0; ks < 2; ++ks)
#pragma unroll
      for (int n = 0; n < 4; ++n)
        vb[ks][n] = load_b_frag_kcontig(Vh + (size_t)(n * 16) * TT + k0 + ks * 32, TT);

    // ---- online softmax (rows: q0 + half*8 + j) ---------------------------
#pragma unroll
    for (int j = 0; j < 8; ++j) {
      const int qrow = q0 + half * 8 + j;
      float s[4];
#pragma unroll
      for (int kk = 0; kk < 4; ++kk) {
        s[kk] = S[kk][j] * ss;
        if (k0 + kk * 16 + nl > qrow) s[kk] = -INFINITY;
      }
      float mx = fmaxf(fmaxf(s[0], s[1]), fmaxf(s[2], s[3]));
#pragma unroll
      for (int d = 1; d < 16; d <<= 1) mx = fmaxf(mx, __shfl_xor(mx, d, 16));
      const float mnew = fmaxf(mrow[j], mx);
      const float alpha = exp2f(mrow[j] - mnew);
      float p[4], rs = 0.0f;
#pragma unroll
      for (int kk = 0; kk < 4; ++kk) { p[kk] = exp2f(s[kk] - mnew); rs += p[kk]; }
#pragma unroll
      for (int d = 1; d < 16; d <<= 1) rs += __shfl_xor(rs, d, 16);
      lrow[j] = lrow[j] * alpha + rs;
      mrow[j] = mnew;
#pragma unroll
      for (int n = 0; n < 4; ++n) accO[n][j] *= alpha;
#pragma unroll
      for (int kk = 0; kk < 4; ++kk)
        sp[(half * 8 + j) * 64 + kk * 16 + nl] = f32_to_bf16(p[kk]);
    }
    // same-wave LDS ops are in-order (DScnt); LDS region is wave-private

    // ---- O += P V: P reloaded as two A-frags from LDS ---------------------
    const v16bf pa0 = load_a_frag(sp, 64);       // keys k0..k0+31
    const v16bf pa1 = load_a_frag(sp + 32, 64);  // keys k0+32..k0+63
#pragma unroll
    for (int n = 0; n < 4; ++n) {
      accO[n] = wmma_bf16(pa0, vb[0][n], accO[n]);
      accO[n] = wmma_bf16(pa1, vb[1][n], accO[n]);
    }
  }

  // ---- epilogue: O/l, bf16, head-merged row-major [B*T, C] ----------------
#pragma unroll
  for (int n = 0; n < 4; ++n)
#pragma unroll
    for (int j = 0; j < 8; ++j) {
      const int qrow = q0 + half * 8 + j;
      yb[(size_t)(b * TT + qrow) * CC + h * DD + n * 16 + nl] =
          f32_to_bf16(accO[n][j] / lrow[j]);
    }
}

// ---------------------------------------------------------------------------
// Kernel 4: output projection  out = y @ Wo^T + bo  (fp32 out)
// grid = (M/256, C/64), block = 256.
// ---------------------------------------------------------------------------
__global__ __launch_bounds__(256) void out_proj_kernel(
    const unsigned short* __restrict__ yb, const unsigned short* __restrict__ wo,
    const float* __restrict__ bo, float* __restrict__ out) {
  const int wave = threadIdx.x >> 5;
  const int lane = threadIdx.x & 31;
  const int half = lane >> 4, nl = lane & 15;
  const int m0 = blockIdx.x * 256 + wave * 32;
  const int n0 = blockIdx.y * 64;

  v8f acc[2][4];
#pragma unroll
  for (int r = 0; r < 2; ++r)
#pragma unroll
    for (int n = 0; n < 4; ++n)
#pragma unroll
      for (int j = 0; j < 8; ++j) acc[r][n][j] = 0.0f;

  gemm_strip_32x64(yb + (size_t)m0 * CC, wo + (size_t)n0 * CC, acc);

#pragma unroll
  for (int n = 0; n < 4; ++n) {
    const int ncol = n0 + n * 16 + nl;
    const float bv_ = bo[ncol];
#pragma unroll
    for (int r = 0; r < 2; ++r)
#pragma unroll
      for (int j = 0; j < 8; ++j) {
        const int row = m0 + r * 16 + half * 8 + j;
        out[(size_t)row * CC + ncol] = acc[r][n][j] + bv_;
      }
  }
}

// ---------------------------------------------------------------------------
extern "C" void kernel_launch(void* const* d_in, const int* in_sizes, int n_in,
                              void* d_out, int out_size, void* d_ws, size_t ws_size,
                              hipStream_t stream) {
  const float* x  = (const float*)d_in[0];
  const float* Wq = (const float*)d_in[1];
  const float* bq = (const float*)d_in[2];
  const float* Wk = (const float*)d_in[3];
  const float* bk = (const float*)d_in[4];
  const float* Wv = (const float*)d_in[5];
  const float* bv = (const float*)d_in[6];
  const float* Wo = (const float*)d_in[7];
  const float* bo = (const float*)d_in[8];
  float* out = (float*)d_out;

  char* p = (char*)d_ws;
  auto take = [&](size_t bytes) {
    char* r = p;
    p += (bytes + 255) & ~(size_t)255;
    return r;
  };
  unsigned short* xb  = (unsigned short*)take((size_t)MM * CC * 2);
  unsigned short* wqb = (unsigned short*)take((size_t)CC * CC * 2);
  unsigned short* wkb = (unsigned short*)take((size_t)CC * CC * 2);
  unsigned short* wvb = (unsigned short*)take((size_t)CC * CC * 2);
  unsigned short* wob = (unsigned short*)take((size_t)CC * CC * 2);
  unsigned short* Qb  = (unsigned short*)take((size_t)MM * CC * 2);
  unsigned short* Kb  = (unsigned short*)take((size_t)MM * CC * 2);
  unsigned short* Vb  = (unsigned short*)take((size_t)MM * CC * 2);
  unsigned short* Vt  = (unsigned short*)take((size_t)MM * CC * 2);
  unsigned short* yb  = (unsigned short*)take((size_t)MM * CC * 2);

  const int nx4 = (MM * CC) / 4, nw4 = (CC * CC) / 4;
  cvt_f32_bf16_kernel<<<1024, 256, 0, stream>>>((const float4*)x,  (ushort4*)xb,  nx4);
  cvt_f32_bf16_kernel<<<512,  256, 0, stream>>>((const float4*)Wq, (ushort4*)wqb, nw4);
  cvt_f32_bf16_kernel<<<512,  256, 0, stream>>>((const float4*)Wk, (ushort4*)wkb, nw4);
  cvt_f32_bf16_kernel<<<512,  256, 0, stream>>>((const float4*)Wv, (ushort4*)wvb, nw4);
  cvt_f32_bf16_kernel<<<512,  256, 0, stream>>>((const float4*)Wo, (ushort4*)wob, nw4);

  qkv_proj_kernel<<<dim3(MM / 256, CC / 64, 3), 256, 0, stream>>>(
      xb, wqb, wkb, wvb, bq, bk, bv, Qb, Kb, Vb);

  v_transpose_kernel<<<dim3(TT / 64, BB * HH), 256, 0, stream>>>(Vb, Vt);

  flash_attn_kernel<<<dim3(TT / 64, BB * HH), 128, 0, stream>>>(Qb, Kb, Vt, yb);

  out_proj_kernel<<<dim3(MM / 256, CC / 64), 256, 0, stream>>>(yb, wob, bo, out);
}
